// AttentionKernel_57019985822087
// MI455X (gfx1250) — matrix operand order repaired and checked by reading, not compiled, not run on hardware
//
#include <hip/hip_runtime.h>
#include <math.h>

// ---------------------------------------------------------------------------
// MI455X (gfx1250) fused neighborhood-attention + edge-MLP.
//
// Roofline: ~4.1 GFLOP total vs ~70 MB of unique HBM traffic -> memory bound
// (~3us at 23.3 TB/s). Strategy:
//  * Two WMMA GEMMs (v_wmma_f32_16x16x32_f16) project K and Q into an 8MB
//    scratch buffer each; that buffer is L2-resident (192MB L2), so the
//    random neighbor gather never materializes the 268MB Kn tensor.
//  * Projection weight panels are DMA'd into LDS by the Tensor Data Mover
//    (tensor_load_to_lds, TENSORcnt) and pre-packed into WMMA-B f16
//    fragments once per block.
//  * One fused kernel: wave per (b,n), lane == m (M=32 == wave32).
//    feat = gathered dot products (VALU, 3% of FLOPs); the per-edge MLP
//    (10->16->16->1, silu) runs as padded 16x16x32 f16 WMMAs per head with
//    all-head B fragments pre-packed in LDS.
// ---------------------------------------------------------------------------

#define B_  2
#define N_  4096
#define M_  32
#define H_  8
#define FD  256
#define LD  10
#define HD  16
#define RTOT (B_ * N_)   // 8192 rows in the projection GEMMs

typedef __attribute__((ext_vector_type(16))) _Float16 v16h;
typedef __attribute__((ext_vector_type(8)))  float    v8f;
typedef __attribute__((ext_vector_type(4)))  unsigned int u32x4;
typedef __attribute__((ext_vector_type(8)))  unsigned int u32x8;

__device__ __forceinline__ float silu_f(float x) {
    return x / (1.0f + __expf(-x));
}

// ---------------------------------------------------------------------------
// Projection: Y[r,c] = sum_k X[r,k] * W[k,c] + bias[c]
// Grid: blockIdx = tc*64 + rblk; the 8 waves of a block handle 8 row tiles of
// the SAME column tile tc, so the 256x16 f32 weight panel is loaded once per
// block via the Tensor Data Mover and pre-packed into WMMA-B f16 fragments.
//
// WMMA fragment layouts (wave32, 16x16x32 f16):
//  A: lanes 0-15 hold row (lane&15): K=0..7 in halves 0..7, K=16..23 in 8..15;
//     lanes 16-31 hold K=8..15 / K=24..31.
//  B: lane n<16 holds column n, K=0..7 / K=16..23 across halves; lanes 16-31 K=8..15 / K=24..31.
//  C: lane l, vgpr i -> (M = i + 8*(l>>4), N = l&15).
// ---------------------------------------------------------------------------
__global__ __launch_bounds__(256) void proj_kernel(
    const float* __restrict__ X, const float* __restrict__ W,
    const float* __restrict__ bias, float* __restrict__ Y)
{
    __shared__ __align__(16) float    sPanel[FD][16];      // W[:, tc*16..+16) 16KB
    __shared__ __align__(32) _Float16 sBfrag[8][32][16];   // packed B frags    8KB

    const int tid  = threadIdx.x;
    const int lane = tid & 31;
    const int warp = tid >> 5;
    const int tc   = blockIdx.x >> 6;               // 0..15  column tile
    const int tr   = (blockIdx.x & 63) * 8 + warp;  // 0..511 row tile

    // --- TDM: DMA the 256x16 f32 weight panel into LDS (wave 0 issues). ---
    // D# group0: count=1, lds_addr, global_addr[56:0], type=2 ("image").
    // D# group1: data_size=4B, tensor 256x256, tile 16x256, dim0 stride 256.
    if (warp == 0) {
        unsigned long long ga = (unsigned long long)(const void*)(W + (size_t)tc * 16);
        unsigned ldsOff = (unsigned)(size_t)(&sPanel[0][0]);  // low 32b = LDS offset
        u32x4 g0;
        g0[0] = 1u;                                        // count = 1
        g0[1] = ldsOff;                                    // lds_addr (bytes)
        g0[2] = (unsigned)(ga & 0xffffffffu);              // global_addr[31:0]
        g0[3] = (unsigned)((ga >> 32) & 0x01ffffffu) | (2u << 30); // [56:32]|type=2
        u32x8 g1;
        g1[0] = 2u << 16;                                  // data_size = 4 bytes
        g1[1] = (unsigned)FD << 16;                        // tensor_dim0 = 256 (lo16)
        g1[2] = (unsigned)FD << 16;                        // tensor_dim1 = 256 (lo16)
        g1[3] = 16u << 16;                                 // tile_dim0 = 16
        g1[4] = (unsigned)FD;                              // tile_dim1 = 256, tile_dim2=0
        g1[5] = (unsigned)FD;                              // tensor_dim0_stride = 256
        g1[6] = 0u;                                        // stride0 hi / stride1 lo16
        g1[7] = 1u;                                        // tensor_dim1_stride = 65536
        asm volatile("tensor_load_to_lds %0, %1" :: "s"(g0), "s"(g1) : "memory");
        __builtin_amdgcn_s_wait_tensorcnt(0);
    }
    __syncthreads();

    // --- Pack panel -> WMMA-B f16 fragments; one thread per (kstep, lane). ---
    {
        const int ks = tid >> 5;                 // k-step 0..7
        const int l  = tid & 31;
        const int nc = l & 15;
        const int kb = ks * 32 + ((l < 16) ? 0 : 8);
        #pragma unroll
        for (int j = 0; j < 16; ++j)
            sBfrag[ks][l][j] = (_Float16)sPanel[kb + j + ((j >= 8) ? 8 : 0)][nc];
    }
    __syncthreads();

    // --- Main WMMA loop: A from global (coalesced b128 + cvt), B from LDS. ---
    const int rowA = tr * 16 + (lane & 15);
    const int aOff = (lane < 16) ? 0 : 8;
    const int colB = tc * 16 + (lane & 15);
    const float* aRow = X + (size_t)rowA * FD;

    v8f c = {};
    #pragma unroll
    for (int k0 = 0; k0 < FD; k0 += 32) {
        float4 x0 = *(const float4*)(aRow + k0 + aOff);
        float4 x1 = *(const float4*)(aRow + k0 + aOff + 4);
        float4 x2 = *(const float4*)(aRow + k0 + aOff + 16);
        float4 x3 = *(const float4*)(aRow + k0 + aOff + 20);

        v16h a;
        a[0]  = (_Float16)x0.x; a[1]  = (_Float16)x0.y;
        a[2]  = (_Float16)x0.z; a[3]  = (_Float16)x0.w;
        a[4]  = (_Float16)x1.x; a[5]  = (_Float16)x1.y;
        a[6]  = (_Float16)x1.z; a[7]  = (_Float16)x1.w;
        a[8]  = (_Float16)x2.x; a[9]  = (_Float16)x2.y;
        a[10] = (_Float16)x2.z; a[11] = (_Float16)x2.w;
        a[12] = (_Float16)x3.x; a[13] = (_Float16)x3.y;
        a[14] = (_Float16)x3.z; a[15] = (_Float16)x3.w;

        v16h b = *(const v16h*)(&sBfrag[k0 >> 5][lane][0]);

        c = __builtin_amdgcn_wmma_f32_16x16x32_f16(
                false, a, false, b, (short)0, c, false, false);
    }

    const float bv    = bias[colB];
    const int   rBase = tr * 16 + ((lane >> 4) << 3);
    #pragma unroll
    for (int i = 0; i < 8; ++i)
        Y[(size_t)(rBase + i) * FD + colB] = c[i] + bv;
}

// ---------------------------------------------------------------------------
// Fused: feat (gathered dots) + edge MLP (WMMA) + combine + store.
// Block = 256 threads = 8 waves; wave w handles one (b,n); lane == m.
// ---------------------------------------------------------------------------
__global__ __launch_bounds__(256) void fused_kernel(
    const float* __restrict__ Kbuf, const float* __restrict__ Qbuf,
    const float* __restrict__ g,    const int*   __restrict__ nidx,
    const float* __restrict__ W0,   const float* __restrict__ b0,
    const float* __restrict__ W1,   const float* __restrict__ b1,
    const float* __restrict__ W2,   const float* __restrict__ b2,
    float* __restrict__ out)
{
    __shared__ __align__(32) _Float16 sB0f[H_][32][16];  // packed W0 B-frags 8KB
    __shared__ __align__(32) _Float16 sB1f[H_][32][16];  // packed W1 B-frags 8KB
    __shared__ float sW2[H_][HD];
    __shared__ float sb0[H_][HD];
    __shared__ float sb1[H_][HD];
    __shared__ float sb2[H_];
    __shared__ __align__(16) float sQ[8][FD];            // 8KB
    __shared__ float sG[8][M_][LD];                      // 10KB
    __shared__ float sHt[8][16][16];                     // 8KB

    const int tid = threadIdx.x;

    // --- Pack per-head B fragments straight from global, once per block. ---
    // thread t: head h = t>>5, fragment lane l = t&31 (256 == 8*32 exactly).
    {
        const int h  = tid >> 5;
        const int l  = tid & 31;
        const int nc = l & 15;
        const int kb = (l < 16) ? 0 : 8;
        #pragma unroll
        for (int j = 0; j < 16; ++j) {
            const int k = kb + j + ((j >= 8) ? 8 : 0);
            sB0f[h][l][j] = (k < LD) ? (_Float16)W0[(h * LD + k) * HD + nc]
                                     : (_Float16)0.0f;
            sB1f[h][l][j] = (k < HD) ? (_Float16)W1[(h * HD + k) * HD + nc]
                                     : (_Float16)0.0f;
        }
    }
    for (int i = tid; i < H_ * HD; i += 256) {
        (&sW2[0][0])[i] = W2[i];
        (&sb0[0][0])[i] = b0[i];
        (&sb1[0][0])[i] = b1[i];
    }
    if (tid < H_) sb2[tid] = b2[tid];

    const int lane = tid & 31;
    const int warp = tid >> 5;
    const int gw   = blockIdx.x * 8 + warp;      // 0 .. B_*N_-1
    const int bb   = gw >> 12;                   // / N_

    // Stage Q row (256 f32) into LDS: 8 floats per lane.
    {
        const float* qp = Qbuf + (size_t)gw * FD;
        *(float4*)(&sQ[warp][lane * 8])     = *(const float4*)(qp + lane * 8);
        *(float4*)(&sQ[warp][lane * 8 + 4]) = *(const float4*)(qp + lane * 8 + 4);
    }
    // Stage g rows: lane == m loads its 10 values.
    {
        const float* gp = g + ((size_t)gw * M_ + lane) * LD;
        #pragma unroll
        for (int j = 0; j < LD; ++j) sG[warp][lane][j] = gp[j];
    }
    const int ki = nidx[(size_t)gw * M_ + lane];

    __syncthreads();

    // ---- feat[m,h] = (1/sqrt(32)) * sum_d Q[h,d] * K[idx[m]][h,d] ----
    float acc[H_] = {};
    const float* kp = Kbuf + ((size_t)bb * N_ + ki) * FD;
    __builtin_prefetch(kp, 0, 1);
    #pragma unroll
    for (int j0 = 0; j0 < FD; j0 += 4) {
        float4 kv = *(const float4*)(kp + j0);
        float4 qv = *(const float4*)(&sQ[warp][j0]);
        acc[j0 >> 5] += kv.x * qv.x + kv.y * qv.y + kv.z * qv.z + kv.w * qv.w;
    }
    #pragma unroll
    for (int h = 0; h < H_; ++h) acc[h] *= 0.17677669529663687f;  // 1/sqrt(32)

    // ---- Build the two A tiles (rows m=0..15, m=16..31) of g, padded to K=32.
    v16h A0[2];
    #pragma unroll
    for (int t = 0; t < 2; ++t) {
        const int row = t * 16 + (lane & 15);
        v16h a = {};
        if (lane < 16) {
            #pragma unroll
            for (int j = 0; j < 8; ++j) a[j] = (_Float16)sG[warp][row][j];
        } else {
            a[0] = (_Float16)sG[warp][row][8];
            a[1] = (_Float16)sG[warp][row][9];
        }
        A0[t] = a;
    }

    const int ncol = lane & 15;

    #pragma unroll 1
    for (int h = 0; h < H_; ++h) {
        const v16h B0v = *(const v16h*)(&sB0f[h][lane][0]);
        const v16h B1v = *(const v16h*)(&sB1f[h][lane][0]);
        const float bias0  = sb0[h][ncol];
        const float bias1  = sb1[h][ncol];
        const float bias2v = sb2[h];

        #pragma unroll 1
        for (int t = 0; t < 2; ++t) {
            // layer 0: h0 = silu(g @ W0[h] + b0[h])
            v8f c0 = {};
            c0 = __builtin_amdgcn_wmma_f32_16x16x32_f16(
                     false, A0[t], false, B0v, (short)0, c0, false, false);
            #pragma unroll
            for (int i = 0; i < 8; ++i)
                sHt[warp][i + ((lane >> 4) << 3)][ncol] = silu_f(c0[i] + bias0);
            __syncthreads();

            // Re-pack h0 from C layout into A layout (K=16, padded to 32).
            v16h A1 = {};
            {
                const int row = lane & 15;
                if (lane < 16) {
                    #pragma unroll
                    for (int j = 0; j < 8; ++j) A1[j] = (_Float16)sHt[warp][row][j];
                } else {
                    #pragma unroll
                    for (int j = 0; j < 8; ++j) A1[j] = (_Float16)sHt[warp][row][8 + j];
                }
            }

            // layer 1: h1 = silu(h0 @ W1[h] + b1[h])
            v8f c1 = {};
            c1 = __builtin_amdgcn_wmma_f32_16x16x32_f16(
                     false, A1, false, B1v, (short)0, c1, false, false);
            #pragma unroll
            for (int i = 0; i < 8; ++i)
                sHt[warp][i + ((lane >> 4) << 3)][ncol] = silu_f(c1[i] + bias1);
            __syncthreads();

            // layer 2: loc = silu(h1 @ W2[h] + b2[h])  (16 -> 1, VALU dot)
            {
                const int row = lane & 15;
                float d = bias2v;
                #pragma unroll
                for (int j = 0; j < HD; ++j) d += sHt[warp][row][j] * sW2[h][j];
                d = silu_f(d);
                if ((lane >> 4) == t) {
                    // global m == lane here (t*16 + row == lane)
                    out[((size_t)gw * M_ + lane) * H_ + h] = acc[h] + d;
                }
            }
            __syncthreads();
        }
    }
}

// ---------------------------------------------------------------------------
extern "C" void kernel_launch(void* const* d_in, const int* in_sizes, int n_in,
                              void* d_out, int out_size, void* d_ws, size_t ws_size,
                              hipStream_t stream) {
    const float* g_pair = (const float*)d_in[0];   // (B,N,M,LD) f32
    // d_in[1] = nbhd_mask (bool) -- unused by the reference output
    const float* keyf   = (const float*)d_in[2];   // (B,N,FD)
    const float* qryf   = (const float*)d_in[3];   // (B,N,FD)
    const int*   idx    = (const int*)  d_in[4];   // (B,N,M) i32
    const float* Wk     = (const float*)d_in[5];
    const float* bk     = (const float*)d_in[6];
    const float* Wq     = (const float*)d_in[7];
    const float* bq     = (const float*)d_in[8];
    const float* W0     = (const float*)d_in[9];
    const float* b0     = (const float*)d_in[10];
    const float* W1     = (const float*)d_in[11];
    const float* b1     = (const float*)d_in[12];
    const float* W2     = (const float*)d_in[13];
    const float* b2     = (const float*)d_in[14];

    float* Kbuf = (float*)d_ws;                    // 8192*256 f32 = 8 MB
    float* Qbuf = Kbuf + (size_t)RTOT * FD;        // 8 MB

    // 16 column tiles * 64 blocks; each block: 8 waves = 8 row tiles, one
    // TDM-loaded weight panel shared by the whole block.
    const int projBlocks = (FD / 16) * ((RTOT / 16) / 8);   // 1024
    proj_kernel<<<projBlocks, 256, 0, stream>>>(keyf, Wk, bk, Kbuf);
    proj_kernel<<<projBlocks, 256, 0, stream>>>(qryf, Wq, bq, Qbuf);

    // 8192 (b,n) instances, 8 per block.
    const int fusedBlocks = RTOT / 8;                       // 1024
    fused_kernel<<<fusedBlocks, 256, 0, stream>>>(
        Kbuf, Qbuf, g_pair, idx, W0, b0, W1, b1, W2, b2, (float*)d_out);
}
